// SimCSEModel_14809047236881
// MI455X (gfx1250) — matrix operand order repaired and checked by reading, not compiled, not run on hardware
//
#include <hip/hip_runtime.h>
#include <hip/hip_bf16.h>
#include <stdint.h>

// ---------- types ----------
typedef __attribute__((ext_vector_type(2)))  float    v2f;
typedef __attribute__((ext_vector_type(8)))  float    v8f;
typedef __attribute__((ext_vector_type(16))) _Float16 v16h;
typedef __attribute__((ext_vector_type(8)))  _Float16 v8h;

#define B_SZ   8192
#define H_SZ   256
#define LMAX   8
#define LOG2E  1.4426950408889634f
#define LN2    0.6931471805599453f
#define INV_T  20.0f                       // 1/0.05
#define SCALE_LOG2E (INV_T * LOG2E)        // 28.85390f

// LDS B-tile: 16 rows x 256 halves, row stride padded to 528 B (264 halves)
// so 16-lane fragment reads start at banks {4n..} -> conflict-free tiling.
#define PAD_HALF 264
#define TILE_LDS_HALF (16 * PAD_HALF)

__device__ __forceinline__ float fast_log(float x)  { return __builtin_amdgcn_logf(x) * LN2; }
__device__ __forceinline__ float fast_tanh(float x) {
    x = fminf(fmaxf(x, -10.f), 10.f);
    float t = __builtin_amdgcn_exp2f(x * (2.0f * LOG2E));
    return (t - 1.0f) * __builtin_amdgcn_rcpf(t + 1.0f);
}

// ---------- kernel 0: zero scalar output + per-row sum accumulators ----------
__global__ void zero_kernel(float* out, float* rowsum) {
    const int i = blockIdx.x * blockDim.x + threadIdx.x;
    if (i < B_SZ) rowsum[i] = 0.0f;
    if (i == 0) out[0] = 0.0f;
}

// ---------- kernel 1: z = tanh(e @ W^T + b), fp32 WMMA 16x16x4 ----------
__global__ __launch_bounds__(256)
void proj_tanh_kernel(const float* __restrict__ e1, const float* __restrict__ e2,
                      const float* __restrict__ W,  const float* __restrict__ bias,
                      float* __restrict__ z /* [2*B, H] */) {
    const int lane = threadIdx.x & 31;
    const int wid  = threadIdx.x >> 5;
    const int tile = blockIdx.x * 8 + wid;          // 0..16383
    const int nt   = tile & 15;
    const int mt   = (tile >> 4) & 511;
    const int mat  = tile >> 13;

    const int m  = lane & 15;
    const int hh = lane >> 4;

    const float* E    = mat ? e2 : e1;
    const float* erow = E + (size_t)(mt * 16 + m) * H_SZ;
    const float* wrow = W + (size_t)(nt * 16 + m) * H_SZ;   // B[k][n] = W[n][k]

    v8f c = {};
    #pragma unroll 8
    for (int k = 0; k < H_SZ; k += 4) {
        v2f a  = *(const v2f*)(erow + k + 2 * hh);
        v2f bb = *(const v2f*)(wrow + k + 2 * hh);
        c = __builtin_amdgcn_wmma_f32_16x16x4_f32(false, a, false, bb,
                                                  (short)0, c, false, false);
    }

    const float bv = bias[nt * 16 + m];
    const size_t rowbase = (size_t)mat * B_SZ + (size_t)mt * 16;
    #pragma unroll
    for (int r = 0; r < 8; ++r) {
        float x = c[r] + bv;
        z[(rowbase + r + (hh ? 8 : 0)) * H_SZ + nt * 16 + m] = fast_tanh(x);
    }
}

// ---------- kernel 2: row-normalize; write f32 (in place) + f16 copy ----------
__global__ __launch_bounds__(256)
void normalize_kernel(float* __restrict__ z /* [2B,H] */, _Float16* __restrict__ h16) {
    const int lane = threadIdx.x & 31;
    const int wid  = threadIdx.x >> 5;
    const size_t row = (size_t)blockIdx.x * 8 + wid;   // 0..16383

    float* p = z + row * H_SZ + lane * 8;
    float4 a = *(const float4*)(p);
    float4 b = *(const float4*)(p + 4);

    float ss = a.x*a.x + a.y*a.y + a.z*a.z + a.w*a.w
             + b.x*b.x + b.y*b.y + b.z*b.z + b.w*b.w;
    #pragma unroll
    for (int m = 16; m >= 1; m >>= 1) ss += __shfl_xor(ss, m, 32);

    const float scale = 1.0f / fmaxf(sqrtf(ss), 1e-8f);
    a.x*=scale; a.y*=scale; a.z*=scale; a.w*=scale;
    b.x*=scale; b.y*=scale; b.z*=scale; b.w*=scale;
    *(float4*)(p)     = a;
    *(float4*)(p + 4) = b;

    v8h hv = { (_Float16)a.x,(_Float16)a.y,(_Float16)a.z,(_Float16)a.w,
               (_Float16)b.x,(_Float16)b.y,(_Float16)b.z,(_Float16)b.w };
    *(v8h*)(h16 + row * H_SZ + lane * 8) = hv;
}

// ---------- kernel 3: fused n1@n2^T (f16 WMMA) + row sum-of-exp ----------
// grid (64, 4): block = 128 anchor rows (wave w -> rows +w*16), 128 column
// tiles per block. B tiles staged to LDS with global_load_async_to_lds_b128,
// double-buffered, pipelined with s_wait_asynccnt.
__global__ __launch_bounds__(256)
void gemm_lse_kernel(const _Float16* __restrict__ h16, float* __restrict__ rowsum) {
    __shared__ __align__(64) _Float16 smem[2 * TILE_LDS_HALF];

    const int tid  = threadIdx.x;
    const int lane = tid & 31;
    const int wid  = tid >> 5;
    const int m    = lane & 15;
    const int hh   = lane >> 4;
    const int rowbase = blockIdx.x * 128 + wid * 16;

    // ---- preload 8 A fragments (16x32 f16) = this wave's 16 rows, K=0..255
    const _Float16* arow = h16 + (size_t)(rowbase + m) * H_SZ;
    v16h A[8];
    #pragma unroll
    for (int kk = 0; kk < 8; ++kk) {
        v8h lo = *(const v8h*)(arow + kk * 32 + hh * 8);
        v8h hi = *(const v8h*)(arow + kk * 32 + 16 + hh * 8);
        A[kk] = __builtin_shufflevector(lo, hi, 0,1,2,3,4,5,6,7,8,9,10,11,12,13,14,15);
    }

    // ---- async staging setup: thread t copies 32 B chunk t of the 8 KB tile
    const uint64_t gbase   = (uint64_t)(uintptr_t)(h16 + (size_t)B_SZ * H_SZ); // n2 f16 base
    const unsigned ldsbase = (unsigned)(uintptr_t)(void*)smem;
    const unsigned lds_chunk = (unsigned)((tid >> 4) * (PAD_HALF * 2) + (tid & 15) * 32);
    const int ntbase = blockIdx.y * 128;

    #define ISSUE_STAGE(buf, nt)                                                   \
        do {                                                                       \
            unsigned _lds = ldsbase + (unsigned)(buf) * (TILE_LDS_HALF * 2) + lds_chunk; \
            unsigned _gof = (unsigned)((nt) * 8192 + tid * 32);                    \
            asm volatile("global_load_async_to_lds_b128 %0, %1, %2\n\t"            \
                         "global_load_async_to_lds_b128 %0, %1, %2 offset:16"      \
                         :: "v"(_lds), "v"(_gof), "s"(gbase) : "memory");          \
        } while (0)

    ISSUE_STAGE(0, ntbase + 0);
    ISSUE_STAGE(1, ntbase + 1);

    float s[8];
    #pragma unroll
    for (int r = 0; r < 8; ++r) s[r] = 0.0f;

    for (int i = 0; i < 128; ++i) {
        if (i + 1 < 128) asm volatile("s_wait_asynccnt 0x2" ::: "memory");
        else             asm volatile("s_wait_asynccnt 0x0" ::: "memory");
        __syncthreads();                                   // tile i staged & visible

        const _Float16* bb = smem + (i & 1) * TILE_LDS_HALF + m * PAD_HALF;
        v8f c = {};
        #pragma unroll
        for (int kk = 0; kk < 8; ++kk) {
            // B 32x16: lane n holds K = kk*32 + hh*8 + {0..7} and kk*32 + 16 + hh*8 + {0..7}
            v8h lo = *(const v8h*)(bb + kk * 32 + hh * 8);
            v8h hi = *(const v8h*)(bb + kk * 32 + 16 + hh * 8);
            v16h bf = __builtin_shufflevector(lo, hi, 0,1,2,3,4,5,6,7,8,9,10,11,12,13,14,15);
            c = __builtin_amdgcn_wmma_f32_16x16x32_f16(false, A[kk], false, bf,
                                                       (short)0, c, false, false);
        }
        // logits bounded in [-20,20] -> single-pass sum of exp is safe in f32
        #pragma unroll
        for (int r = 0; r < 8; ++r)
            s[r] += __builtin_amdgcn_exp2f(c[r] * SCALE_LOG2E);

        __syncthreads();                                   // everyone done with buf (i&1)
        if (i + 2 < 128) ISSUE_STAGE(i & 1, ntbase + i + 2);
    }
    #undef ISSUE_STAGE

    // C layout: reg r, lanes 0-15 -> row r; lanes 16-31 -> row r+8.
    #pragma unroll
    for (int r = 0; r < 8; ++r) {
        float v = s[r];
        #pragma unroll
        for (int msk = 8; msk >= 1; msk >>= 1) v += __shfl_xor(v, msk, 32);
        s[r] = v;
    }
    if ((lane & 15) == 0) {
        const int base = rowbase + hh * 8;
        #pragma unroll
        for (int r = 0; r < 8; ++r) atomicAdd(&rowsum[base + r], s[r]); // global_atomic_add_f32
    }
}

// ---------- kernel 4: ragged-label CE + scalar reduction ----------
__global__ __launch_bounds__(256)
void loss_kernel(const float* __restrict__ n /* [2B,H] normalized f32 */,
                 const float* __restrict__ rowsum,
                 const int* __restrict__ labels, const int* __restrict__ counts,
                 float* __restrict__ out) {
    const int lane = threadIdx.x & 31;
    const int wid  = threadIdx.x >> 5;
    const int row  = blockIdx.x * 8 + wid;          // 0..8191

    const float* arow = n + (size_t)row * H_SZ + lane * 8;
    float4 a0 = *(const float4*)(arow);
    float4 a1 = *(const float4*)(arow + 4);

    int cnt = counts[row];
    cnt = min(max(cnt, 1), LMAX);
    const float lse = fast_log(rowsum[row]);

    float dsum = 0.0f;
    for (int j = 0; j < cnt; ++j) {
        const int l = labels[row * LMAX + j];
        const float* brow = n + (size_t)(B_SZ + l) * H_SZ + lane * 8;
        float4 b0 = *(const float4*)(brow);
        float4 b1 = *(const float4*)(brow + 4);
        float d = a0.x*b0.x + a0.y*b0.y + a0.z*b0.z + a0.w*b0.w
                + a1.x*b1.x + a1.y*b1.y + a1.z*b1.z + a1.w*b1.w;
        #pragma unroll
        for (int m = 16; m >= 1; m >>= 1) d += __shfl_xor(d, m, 32);
        dsum += d;
    }
    const float fc = (float)cnt;
    const float contrib = (fc * lse - INV_T * dsum) / (fc * fc);
    if (lane == 0) atomicAdd(out, contrib);
}

// ---------- launch ----------
extern "C" void kernel_launch(void* const* d_in, const int* in_sizes, int n_in,
                              void* d_out, int out_size, void* d_ws, size_t ws_size,
                              hipStream_t stream) {
    const float* e1     = (const float*)d_in[0];
    const float* e2     = (const float*)d_in[1];
    const float* W      = (const float*)d_in[2];
    const float* bias   = (const float*)d_in[3];
    const int*   labels = (const int*)d_in[4];
    const int*   counts = (const int*)d_in[5];
    float*       out    = (float*)d_out;

    // ws: z/n f32 [2B,H] (16 MB) | f16 copy [2B,H] (8 MB) | rowsum [B] f32
    char* ws   = (char*)d_ws;
    float*     z32    = (float*)ws;
    _Float16*  h16    = (_Float16*)(ws + (size_t)2 * B_SZ * H_SZ * sizeof(float));
    float*     rowsum = (float*)(ws + (size_t)2 * B_SZ * H_SZ * (sizeof(float) + sizeof(_Float16)));

    zero_kernel<<<32, 256, 0, stream>>>(out, rowsum);
    proj_tanh_kernel<<<2048, 256, 0, stream>>>(e1, e2, W, bias, z32);
    normalize_kernel<<<2048, 256, 0, stream>>>(z32, h16);
    gemm_lse_kernel<<<dim3(64, 4), 256, 0, stream>>>(h16, rowsum);
    loss_kernel<<<1024, 256, 0, stream>>>(z32, rowsum, labels, counts, out);
}
